// LiquidMalwareDetector_6390911336625
// MI455X (gfx1250) — matrix operand order repaired and checked by reading, not compiled, not run on hardware
//
#include <hip/hip_runtime.h>
#include <hip/hip_bf16.h>
#include <math.h>

// Problem constants (match reference)
#define B_      256
#define T_      1024
#define F_      128
#define H_      512
#define K_      640     // F + H
#define GROUPS  16      // B_/16 batch tiles (independent recurrences)
#define WG_PER_G 16     // N-split of the 1024 output cols -> 32 w1 + 32 w2 cols per WG
#define THREADS 128     // 4 waves (wave32) per workgroup
#define WAVES   4
#define KT      20      // K_/32 k-tiles per WMMA chain
#define TILE_DW 272     // bank-swizzled 32x16 bf16 B-tile footprint (dwords)
#define SLICE_DW (4 * KT * TILE_DW)   // 21760 dwords = 87,040 B per WG weight slice
#define AROW_DW 324     // padded combined-activation row stride (dwords)
#define AROW_US 648

typedef __attribute__((ext_vector_type(16))) __bf16 v16bf;
typedef __attribute__((ext_vector_type(8)))  float  v8f;
typedef __attribute__((ext_vector_type(4)))  unsigned int v4u;
typedef __attribute__((ext_vector_type(8)))  int v8i;
typedef __attribute__((ext_vector_type(4)))  int v4i;

#if defined(__has_builtin)
#  if __has_builtin(__builtin_amdgcn_tensor_load_to_lds) && \
      __has_builtin(__builtin_amdgcn_s_wait_tensorcnt)
#    define HAVE_TDM 1
#  endif
#endif
#ifndef HAVE_TDM
#  define HAVE_TDM 0
#endif

#if defined(__has_builtin) && __has_builtin(__builtin_amdgcn_tanhf)
#  define TANHF(x) __builtin_amdgcn_tanhf(x)
#else
#  define TANHF(x) tanhf(x)
#endif

union Frag { unsigned int u[8]; v16bf v; };

__device__ __forceinline__ unsigned short f2bf(float f) {
  union { float f; unsigned int u; } v; v.f = f;
  unsigned int r = v.u + 0x7fffu + ((v.u >> 16) & 1u);   // round-to-nearest-even
  return (unsigned short)(r >> 16);
}
__device__ __forceinline__ float bf2f(unsigned short h) {
  union { float f; unsigned int u; } v; v.u = ((unsigned int)h) << 16;
  return v.f;
}

#if HAVE_TDM
// ---------------------------------------------------------------------------
// Tensor Data Mover: 2-D tile load (ISA 8.3/8.4 D# bitfields).
//   group0: [1:0]=count=1 | lds_addr[63:32] | global_addr[120:64] | type=2
//   group1: data_size[17:16] | tensor_dim0[79:48] | tensor_dim1[111:80] |
//           tile_dim0[127:112] | tile_dim1[143:128] | tensor_dim0_stride[207:160]
// dims/strides in data_size units; trailing descriptor groups zero (<=2-D).
// This toolchain exposes the 6-arg builtin: (v4u, v8i, v4i, v4i, v8i, cpol).
// ---------------------------------------------------------------------------
__device__ __forceinline__ void tdm_load_2d(unsigned lds_byte_addr,
                                            const void* gptr, int dsz_code,
                                            unsigned dim0, unsigned tile0,
                                            unsigned dim1, unsigned tile1,
                                            unsigned long long stride0) {
  unsigned long long ga = (unsigned long long)(size_t)gptr;
  v4u g0 = { 1u, lds_byte_addr, (unsigned)ga,
             (unsigned)(ga >> 32) | 0x80000000u };
  v8i g1 = { (int)(dsz_code << 16),
             (int)((dim0 & 0xffffu) << 16),
             (int)(((dim0 >> 16) & 0xffffu) | ((dim1 & 0xffffu) << 16)),
             (int)(((dim1 >> 16) & 0xffffu) | (tile0 << 16)),
             (int)(tile1 & 0xffffu),
             (int)(unsigned)stride0,
             (int)((stride0 >> 32) & 0xffffu),
             0 };
  v4i z4 = { 0, 0, 0, 0 };
  v8i z8 = { 0, 0, 0, 0, 0, 0, 0, 0 };
  __builtin_amdgcn_tensor_load_to_lds(g0, g1, z4, z4, z8, 0);
}
#endif

// ---------------------------------------------------------------------------
// Kernel 0: reset the per-group barrier counters (idempotent per call)
// ---------------------------------------------------------------------------
__global__ void liquid_init(unsigned int* __restrict__ cnt) {
  if (threadIdx.x < GROUPS) cnt[threadIdx.x] = 0u;
}

// ---------------------------------------------------------------------------
// Kernel 1: pack W = [w1 ; w2]^T fp32 -> bf16 directly into the bank-swizzled
// LDS image (per-lane WMMA B-fragment order), so the device copy is a straight
// linear DMA. Layout per WG slot j: [ntile w:4][ktile:20][TILE_DW dwords]
// with dword = tile*272 + lane*8 + (lane>>3)*4 + v.
// ---------------------------------------------------------------------------
__global__ void liquid_pack(const float* __restrict__ w1,
                            const float* __restrict__ w2,
                            unsigned int* __restrict__ pw) {
  int idx = blockIdx.x * 256 + threadIdx.x;
  const int TOTAL = WG_PER_G * 4 * KT * 32 * 8;            // 327680 payload dwords
  if (idx >= TOTAL) return;
  int v    = idx & 7;
  int t1   = idx >> 3;
  int lane = t1 & 31;
  int t2   = t1 >> 5;
  int kt   = t2 % KT;
  int t3   = t2 / KT;
  int w    = t3 & 3;        // local n-tile: 0,1 -> w1 ; 2,3 -> w2
  int j    = t3 >> 2;       // workgroup slot in group
  const float* src = (w >> 1) ? w2 : w1;                   // [H][K_] row-major
  int c  = j * 32 + (w & 1) * 16 + (lane & 15);
  int k0 = kt * 32 + ((v >> 2) << 4) + ((lane >> 4) << 3) + 2 * (v & 3);
  unsigned int d = (unsigned int)f2bf(src[c * K_ + k0]) |
                   ((unsigned int)f2bf(src[c * K_ + k0 + 1]) << 16);
  pw[j * SLICE_DW + (w * KT + kt) * TILE_DW + lane * 8 + ((lane >> 3) << 2) + v] = d;
}

// ---------------------------------------------------------------------------
// Inter-workgroup barrier for the 16 WGs of one batch group.
// ---------------------------------------------------------------------------
__device__ __forceinline__ void group_barrier(unsigned int* cnt, unsigned int target) {
  __threadfence();
  __syncthreads();
  if (threadIdx.x == 0) {
    __hip_atomic_fetch_add(cnt, 1u, __ATOMIC_RELEASE, __HIP_MEMORY_SCOPE_AGENT);
    while (__hip_atomic_load(cnt, __ATOMIC_ACQUIRE, __HIP_MEMORY_SCOPE_AGENT) < target) {
      __builtin_amdgcn_s_sleep(1);
    }
  }
  __syncthreads();
}

// ---------------------------------------------------------------------------
// Kernel 2: persistent recurrence. 256 WGs x 128 threads (4 waves).
//   group g = blockIdx>>4 owns batch rows [16g,16g+16)
//   slot  j = blockIdx&15 owns output cols [32j,32j+32) of w1 AND w2
// Weights DMA'd into LDS once (TDM) and reused for all 1024 steps; x tiles
// double-buffered via TDM overlapped with compute; h ping-pongs through L2
// with scoped atomics (read-only WGP$ can never serve stale lines).
// ---------------------------------------------------------------------------
__launch_bounds__(THREADS, 1)
__global__ void liquid_rnn(const float* __restrict__ x,
                           const float* __restrict__ times,
                           const float* __restrict__ b1,
                           const float* __restrict__ b2,
                           const float* __restrict__ wa,
                           const float* __restrict__ ba,
                           const float* __restrict__ wb,
                           const float* __restrict__ bb,
                           const float* __restrict__ wc,
                           const float* __restrict__ bc,
                           const unsigned int* __restrict__ pw,
                           unsigned short* __restrict__ hbuf,   // [2][GROUPS][16][H_]
                           unsigned int* __restrict__ cnt,      // [GROUPS]
                           float* __restrict__ out) {
  __shared__ unsigned int lds_w[SLICE_DW];           // 87,040 B : resident B-fragments
  __shared__ unsigned int lds_a[16 * AROW_DW];       // 20,736 B : combined [x|h] bf16
  __shared__ float        lds_e[2 * 32 * 8];         //  2,048 B : ff2 exchange tiles
  __shared__ float        lds_t[16];                 // per-row time values
#if HAVE_TDM
  __shared__ float        lds_xf[2 * 16 * F_];       // 16,384 B : x tile double buffer
#endif

  const int tid  = threadIdx.x;
  const int g    = blockIdx.x >> 4;
  const int j    = blockIdx.x & 15;
  const int wave = tid >> 5;
  const int lane = tid & 31;

#if HAVE_TDM
  // ---- DMA this WG's weight slice into LDS: each wave moves one quarter.
  {
    unsigned wbase = (unsigned)(size_t)(void*)lds_w;
    const unsigned qdw = SLICE_DW / 4;               // 5440 dwords per wave
    tdm_load_2d(wbase + wave * (qdw * 4), pw + (size_t)j * SLICE_DW + wave * qdw,
                /*dsz=8B*/3, qdw / 2, qdw / 2, 1, 1, qdw / 2);
    // ---- DMA x tile for t=0: each wave moves 4 rows of [16 x 128] fp32.
    unsigned xbase = (unsigned)(size_t)(void*)lds_xf;
    tdm_load_2d(xbase + wave * (4 * F_ * 4),
                x + (((size_t)(g * 16 + wave * 4)) * T_ + 0) * F_,
                /*dsz=4B*/2, F_, F_, 4, 4, (unsigned long long)T_ * F_);
  }
#else
  for (int i = tid; i < SLICE_DW; i += THREADS)
    lds_w[i] = pw[(size_t)j * SLICE_DW + i];
#endif

  // ---- per-lane column constants (each wave owns one fixed 16-col N tile)
  const int   cl   = j * 32 + (wave & 1) * 16 + (lane & 15);
  const float bias = (wave < 2) ? b1[cl] : b2[cl];
  const float wac = wa[cl], bac = ba[cl], wbc = wb[cl], bbc = bb[cl];

  // ---- zero our slice of h buffer 0 (h0 = 0), device-visible
  {
    unsigned long long* hz =
        (unsigned long long*)(hbuf + ((size_t)(0 * GROUPS + g) * 16) * H_);
    if (tid < 128) {
      int m = tid >> 3, c8 = tid & 7;
      __hip_atomic_store(&hz[(m * H_ + j * 32 + c8 * 4) >> 2], 0ull,
                         __ATOMIC_RELAXED, __HIP_MEMORY_SCOPE_AGENT);
    }
  }
  unsigned int rnd = 1;
  group_barrier(&cnt[g], WG_PER_G * rnd); ++rnd;

  const unsigned short* lds_a_us = (const unsigned short*)lds_a;
  const unsigned short* a_base   = lds_a_us + (lane & 15) * AROW_US + ((lane >> 4) << 3);
  const unsigned int*   b_base   = lds_w + wave * KT * TILE_DW + lane * 8 + ((lane >> 3) << 2);

  for (int t = 0; t < T_; ++t) {
    // -- stage h_t (scoped loads force-miss WGP$) into combined[:,128:640)
    const unsigned long long* hsrc = (const unsigned long long*)
        (hbuf + ((size_t)((t & 1) * GROUPS + g) * 16) * H_);
    for (int q = tid; q < 2048; q += THREADS) {
      unsigned long long d =
          __hip_atomic_load(&hsrc[q], __ATOMIC_RELAXED, __HIP_MEMORY_SCOPE_AGENT);
      int m = q >> 7, p = q & 127;
      lds_a[m * AROW_DW + 64 + p * 2]     = (unsigned int)d;
      lds_a[m * AROW_DW + 64 + p * 2 + 1] = (unsigned int)(d >> 32);
    }

#if HAVE_TDM
    // -- overlap: kick off DMA of x tile t+1, then wait for tile t (in-order
    //    per-wave TENSORcnt, so <=1 leaves next tile's DMA in flight).
    if (t + 1 < T_) {
      unsigned xbase = (unsigned)(size_t)(void*)lds_xf;
      tdm_load_2d(xbase + ((t + 1) & 1) * (16 * F_ * 4) + wave * (4 * F_ * 4),
                  x + (((size_t)(g * 16 + wave * 4)) * T_ + (t + 1)) * F_,
                  2, F_, F_, 4, 4, (unsigned long long)T_ * F_);
      __builtin_amdgcn_s_wait_tensorcnt(1);
    } else {
      __builtin_amdgcn_s_wait_tensorcnt(0);
    }
    __syncthreads();   // all waves' DMA quarters of tile t now visible
    // -- convert staged x tile fp32 -> bf16 into combined[:,0:128)
    {
      const float* xf = &lds_xf[(t & 1) * (16 * F_)];
      for (int i = tid; i < 1024; i += THREADS) {
        int m = i >> 6, p = i & 63;
        lds_a[m * AROW_DW + p] =
            (unsigned int)f2bf(xf[m * F_ + p * 2]) |
            ((unsigned int)f2bf(xf[m * F_ + p * 2 + 1]) << 16);
      }
    }
#else
    for (int i = tid; i < 1024; i += THREADS) {
      int m = i >> 6, p = i & 63;
      const float* xs = x + (((size_t)(g * 16 + m)) * T_ + t) * F_ + p * 2;
      lds_a[m * AROW_DW + p] =
          (unsigned int)f2bf(xs[0]) | ((unsigned int)f2bf(xs[1]) << 16);
    }
#endif
    if (tid < 16) lds_t[tid] = times[((size_t)(g * 16 + tid)) * T_ + t];
    __syncthreads();

    // -- WMMA chain over K=640: D = combined @ Wslice + bias
    v8f acc;
#pragma unroll
    for (int i = 0; i < 8; ++i) acc[i] = bias;
#pragma unroll
    for (int kt = 0; kt < KT; ++kt) {
      Frag fa, fb;
      *(uint4*)&fa.u[0] = *(const uint4*)(const void*)(a_base + kt * 32);
      *(uint4*)&fa.u[4] = *(const uint4*)(const void*)(a_base + kt * 32 + 16);
      *(uint4*)&fb.u[0] = *(const uint4*)(const void*)(b_base + kt * TILE_DW);
      *(uint4*)&fb.u[4] = *(const uint4*)(const void*)(b_base + kt * TILE_DW + 4);
      acc = __builtin_amdgcn_wmma_f32_16x16x32_bf16(
          false, fa.v, false, fb.v, (short)0, acc, false, false);
    }
#pragma unroll
    for (int i = 0; i < 8; ++i) acc[i] = TANHF(acc[i]);   // native V_TANH_F32

    // -- waves 2,3 export tanh(ff2) tiles through LDS to their partner waves
    if (wave >= 2) {
      float* e = &lds_e[((wave - 2) * 32 + lane) * 8];
#pragma unroll
      for (int i = 0; i < 8; ++i) e[i] = acc[i];
    }
    __syncthreads();

    // -- waves 0,1: gate + blend, emit h_{t+1} slice (bf16) to the other buffer
    if (wave < 2) {
      unsigned short* hdst =
          hbuf + ((size_t)(((t + 1) & 1) * GROUPS + g) * 16) * H_;
      const float* e = &lds_e[(wave * 32 + lane) * 8];
#pragma unroll
      for (int r = 0; r < 8; ++r) {
        int   m   = r + ((lane >> 4) << 3);         // C-matrix row for this VGPR
        float tt  = lds_t[m];
        float zg  = (tt * wac + bac) * tt + (tt * wbc + bbc);
        float gte = 0.5f + 0.5f * TANHF(0.5f * zg); // sigmoid via native tanh
        float nh  = acc[r] * gte + e[r] * (1.0f - gte);
        hdst[m * H_ + cl] = f2bf(nh);
      }
    }
    group_barrier(&cnt[g], WG_PER_G * rnd); ++rnd;
  }

  // ---- epilogue: logits + softmax for this group's 16 rows (slot 0 only)
  if (j == 0 && tid < 32) {
    int m = tid >> 1, cls = tid & 1;
    const unsigned short* h =
        hbuf + ((size_t)((T_ & 1) * GROUPS + g) * 16) * H_;   // T_ even -> buf 0
    const float* wcr = wc + cls * H_;
    float s = bc[cls];
    for (int k = 0; k < H_; k += 4) {
      unsigned long long d = __hip_atomic_load(
          (const unsigned long long*)(const void*)(h + m * H_ + k),
          __ATOMIC_RELAXED, __HIP_MEMORY_SCOPE_AGENT);
      s += bf2f((unsigned short)(d      )) * wcr[k]
         + bf2f((unsigned short)(d >> 16)) * wcr[k + 1]
         + bf2f((unsigned short)(d >> 32)) * wcr[k + 2]
         + bf2f((unsigned short)(d >> 48)) * wcr[k + 3];
    }
    float other = __shfl_xor(s, 1, 32);
    float mx = fmaxf(s, other);
    float e0 = __expf(s - mx), e1 = __expf(other - mx);
    out[((size_t)(g * 16 + m)) * 2 + cls] = e0 / (e0 + e1);
  }
}

// ---------------------------------------------------------------------------
// Launch: init counters -> pack weights -> persistent recurrence.
// Workspace map: [0,64)            barrier counters (padded to 256B)
//                [256, +1,392,640) packed, pre-swizzled bf16 weights
//                [.., +524,288)    h double buffer [2][16][16][512] bf16
// ---------------------------------------------------------------------------
extern "C" void kernel_launch(void* const* d_in, const int* in_sizes, int n_in,
                              void* d_out, int out_size, void* d_ws, size_t ws_size,
                              hipStream_t stream) {
  const float* x     = (const float*)d_in[0];
  const float* times = (const float*)d_in[1];
  const float* w1    = (const float*)d_in[2];
  const float* b1    = (const float*)d_in[3];
  const float* w2    = (const float*)d_in[4];
  const float* b2    = (const float*)d_in[5];
  const float* wa    = (const float*)d_in[6];
  const float* ba    = (const float*)d_in[7];
  const float* wb    = (const float*)d_in[8];
  const float* bb    = (const float*)d_in[9];
  const float* wc    = (const float*)d_in[10];
  const float* bc    = (const float*)d_in[11];

  unsigned int*   cnt  = (unsigned int*)d_ws;
  unsigned int*   pw   = (unsigned int*)((char*)d_ws + 256);
  unsigned short* hbuf = (unsigned short*)((char*)d_ws + 256 + (size_t)WG_PER_G * SLICE_DW * 4);
  float*          out  = (float*)d_out;

  liquid_init<<<1, 32, 0, stream>>>(cnt);
  liquid_pack<<<1280, 256, 0, stream>>>(w1, w2, pw);
  liquid_rnn<<<GROUPS * WG_PER_G, THREADS, 0, stream>>>(
      x, times, b1, b2, wa, ba, wb, bb, wc, bc, pw, hbuf, cnt, out);
}
